// RNN_41145786695778
// MI455X (gfx1250) — matrix operand order repaired and checked by reading, not compiled, not run on hardware
//
#include <hip/hip_runtime.h>
#include <hip/hip_bf16.h>
#include <math.h>

// ---------------------------------------------------------------------------
// SimpleRNN on gfx1250.
//   Key fact: batches are independent recurrences (h[b,:] depends only on
//   h_prev[b,:]), so NO inter-workgroup sync is ever needed.
//   Phase 1: xw = x @ W_in + b  -> d_out   (bandwidth-bound bf16-WMMA GEMM)
//   Phase 2: 4 persistent WGs, 16 batch rows each, h ping-pong in LDS,
//            W_rec pre-swizzled to B-fragment order in global (L2-hot),
//            next step's xw tile staged via global_load_async_to_lds_b128.
// ---------------------------------------------------------------------------

typedef __bf16 bf16_t;
typedef __attribute__((ext_vector_type(16))) __bf16 v16bf;
typedef __attribute__((ext_vector_type(8)))  __bf16 v8bf;
typedef __attribute__((ext_vector_type(8)))  float  v8f;
typedef __attribute__((ext_vector_type(4)))  float  v4f;

#define BATCH 64
#define SEQ   512
#define IN_DIM 512
#define UNITS  512
#define ROWS  (BATCH * SEQ)   // 32768

__device__ __forceinline__ v8f wmma_bf16(v16bf a, v16bf b, v8f c) {
  return __builtin_amdgcn_wmma_f32_16x16x32_bf16(false, a, false, b, (short)0,
                                                 c, false, false);
}

// A fragment (16x32 bf16, row-major, stride in elements).
// lanes 0-15 -> M=lane, elems 0..15 = K0..15;
// lanes 16-31 -> M=lane-16, elems 0..15 = K16..31.
__device__ __forceinline__ v16bf load_a_frag(const bf16_t* base, int stride) {
  int lane = threadIdx.x & 31;
  int m  = lane & 15;
  int kb = (lane >> 4) << 4;           // 0 or 16
  const bf16_t* p0 = base + m * stride + kb;
  v8bf lo = *(const v8bf*)p0;          // K = kb .. kb+7
  v8bf hi = *(const v8bf*)(p0 + 8);    // K = kb+8 .. kb+15
  v16bf a;
#pragma unroll
  for (int i = 0; i < 8; ++i) { a[i] = lo[i]; a[8 + i] = hi[i]; }
  return a;
}

// B fragment from a pre-swizzled block: [lane][16 contiguous bf16].
__device__ __forceinline__ v16bf load_b_frag_sw(const bf16_t* blk) {
  int lane = threadIdx.x & 31;
  return *(const v16bf*)(blk + lane * 16);
}

// ---------------------------------------------------------------------------
// Init: W_in -> bf16 (row-major); W_rec -> bf16 pre-swizzled B-fragment order
//   WrecSw layout: [kc(16)][nt(32)][lane(32)][j(16)]
//   source: k = kc*32 + ((lane>>4)<<4) + j ; n = nt*16 + (lane&15)
// ---------------------------------------------------------------------------
__global__ __launch_bounds__(256) void rnn_init(const float* __restrict__ Win,
                                                const float* __restrict__ Wrec,
                                                bf16_t* __restrict__ Winb,
                                                bf16_t* __restrict__ WrecSw) {
  int idx = blockIdx.x * 256 + threadIdx.x;
  if (idx < IN_DIM * UNITS) {
    Winb[idx] = (bf16_t)Win[idx];
    int j  = idx & 15;
    int l  = (idx >> 4) & 31;
    int nt = (idx >> 9) & 31;
    int kc = idx >> 14;
    int k  = kc * 32 + ((l >> 4) << 4) + j;
    int n  = nt * 16 + (l & 15);
    WrecSw[idx] = (bf16_t)Wrec[k * UNITS + n];
  }
}

// ---------------------------------------------------------------------------
// Phase 1: out[r, :] = x[r, :] @ W_in + b  (r = b*SEQ + t), bf16 WMMA.
// Block tile 128(M) x 64(N), 8 waves (4x2), wave tile 32x32 (2x2 C-tiles).
// ---------------------------------------------------------------------------
#define BM 128
#define BN 64
#define BK 32
#define SA (BK + 8)

__global__ __launch_bounds__(256) void rnn_gemm_in(
    const float* __restrict__ x, const bf16_t* __restrict__ Wb,
    const float* __restrict__ bias, float* __restrict__ out) {
  __shared__ bf16_t As[BM * SA];
  __shared__ bf16_t Bs[4 * 32 * 16];

  int tid  = threadIdx.x;
  int wave = tid >> 5;
  int lane = tid & 31;
  int wm = wave >> 1;
  int wn = wave & 1;
  int wg_n = blockIdx.x & 7;
  int wg_m = blockIdx.x >> 3;
  int row0 = wg_m * BM;
  int col0 = wg_n * BN;

  v8f c00, c01, c10, c11;
#pragma unroll
  for (int r = 0; r < 8; ++r) { c00[r] = 0.f; c01[r] = 0.f; c10[r] = 0.f; c11[r] = 0.f; }

  for (int kc = 0; kc < IN_DIM; kc += BK) {
#pragma unroll
    for (int i = 0; i < 4; ++i) {
      int idx = (tid + i * 256) * 4;
      int r = idx >> 5;
      int cidx = idx & 31;
      v4f v = *(const v4f*)(x + (row0 + r) * IN_DIM + kc + cidx);
      bf16_t* d = &As[r * SA + cidx];
      d[0] = (bf16_t)v[0]; d[1] = (bf16_t)v[1];
      d[2] = (bf16_t)v[2]; d[3] = (bf16_t)v[3];
    }
#pragma unroll
    for (int i = 0; i < 8; ++i) {
      int flat = tid * 8 + i;
      int j  = flat & 15;
      int l  = (flat >> 4) & 31;
      int nt = flat >> 9;
      int k  = ((l >> 4) << 4) + j;
      int n  = (nt << 4) + (l & 15);
      Bs[flat] = Wb[(kc + k) * UNITS + col0 + n];
    }
    __syncthreads();

    v16bf a0 = load_a_frag(&As[(wm * 32 + 0)  * SA], SA);
    v16bf a1 = load_a_frag(&As[(wm * 32 + 16) * SA], SA);
    v16bf b0 = load_b_frag_sw(&Bs[(wn * 2 + 0) * 512]);
    v16bf b1 = load_b_frag_sw(&Bs[(wn * 2 + 1) * 512]);
    c00 = wmma_bf16(a0, b0, c00);
    c01 = wmma_bf16(a0, b1, c01);
    c10 = wmma_bf16(a1, b0, c10);
    c11 = wmma_bf16(a1, b1, c11);
    __syncthreads();
  }

  int lm = (lane >> 4) << 3;
  int ln = lane & 15;
#pragma unroll
  for (int i = 0; i < 2; ++i) {
#pragma unroll
    for (int j = 0; j < 2; ++j) {
      v8f* cp = (i == 0) ? (j == 0 ? &c00 : &c01) : (j == 0 ? &c10 : &c11);
      int crow = row0 + wm * 32 + i * 16 + lm;
      int ccol = col0 + wn * 32 + j * 16 + ln;
      float bv = bias[ccol];
#pragma unroll
      for (int r = 0; r < 8; ++r)
        out[(crow + r) * UNITS + ccol] = (*cp)[r] + bv;
    }
  }
}

// ---------------------------------------------------------------------------
// Phase 2: 4 WGs x 1024 threads. WG g owns batch rows [g*16, g*16+16).
// Wave w (0..31) owns output N-tile w. h ping-pong lives in LDS.
// xw tile for step t+1 staged with global_load_async_to_lds_b128 during step t.
// ---------------------------------------------------------------------------
#define HSTRIDE 520   // bf16 elems per h row (padded)
#define XSTRIDE 516   // f32 elems per xw row (padded)

__device__ __forceinline__ void stage_xw_async(float* xs, const float* out,
                                               int row0, int t, int tid) {
  // 16 rows x 512 f32 = 2048 x b128 chunks; 1024 threads -> 2 each.
#pragma unroll
  for (int c = tid; c < 2048; c += 1024) {
    int row = c >> 7;            // 128 chunks per row
    int col = (c & 127) << 2;    // 4 floats per chunk
    const float* gp = out + ((size_t)(row0 + row) * SEQ + t) * UNITS + col;
    uint32_t la = (uint32_t)(uintptr_t)(xs + row * XSTRIDE + col);
    asm volatile("global_load_async_to_lds_b128 %0, %1, off"
                 :: "v"(la), "v"(gp) : "memory");
  }
}

__global__ __launch_bounds__(1024) void rnn_scan(
    const bf16_t* __restrict__ WrecSw, float* out) {
  __shared__ bf16_t Hs[2][16 * HSTRIDE];   // h_{t-1}/h_t ping-pong (bf16)
  __shared__ float  Xs[2][16 * XSTRIDE];   // staged xw tiles (f32)

  int g    = blockIdx.x;                   // 0..3
  int tid  = threadIdx.x;                  // 0..1023
  int wave = tid >> 5;                     // N-tile 0..31
  int lane = tid & 31;
  int lm = (lane >> 4) << 3;
  int ln = lane & 15;
  int row0 = g * 16;
  int ncol = wave * 16 + ln;               // this lane's output unit

  // h_0 = 0
  for (int i = tid; i < 16 * HSTRIDE; i += 1024) Hs[0][i] = (bf16_t)0.0f;

  // Stage xw for t = 0.
  stage_xw_async(&Xs[0][0], out, row0, 0, tid);
  asm volatile("s_wait_asynccnt 0x0" ::: "memory");
  __syncthreads();

  for (int t = 0; t < SEQ; ++t) {
    int cur = t & 1, nxt = cur ^ 1;

    // Kick off async staging of next timestep's xw tile.
    if (t + 1 < SEQ)
      stage_xw_async(&Xs[nxt][0], out, row0, t + 1, tid);

    // Accumulator = staged xw (preactivation, f32).
    v8f acc;
#pragma unroll
    for (int r = 0; r < 8; ++r)
      acc[r] = Xs[cur][(lm + r) * XSTRIDE + ncol];

    // Recurrent GEMM: K = 512 in chunks of 32. A from LDS h, B from global
    // pre-swizzled W_rec (contiguous 32B per lane, L2-resident).
#pragma unroll 2
    for (int kc = 0; kc < 16; ++kc) {
      v16bf a = load_a_frag(&Hs[cur][kc * 32], HSTRIDE);
      v16bf b = *(const v16bf*)(WrecSw + (((kc * 32 + wave) * 32) + lane) * 16);
      acc = wmma_bf16(a, b, acc);
    }

    // tanh; f32 state to out, bf16 state to LDS for next step.
#pragma unroll
    for (int r = 0; r < 8; ++r) {
      float h = tanhf(acc[r]);
      out[((size_t)(row0 + lm + r) * SEQ + t) * UNITS + ncol] = h;
      Hs[nxt][(lm + r) * HSTRIDE + ncol] = (bf16_t)h;
    }

    // Our async stage must be complete, and all waves' h writes visible.
    asm volatile("s_wait_asynccnt 0x0" ::: "memory");
    __syncthreads();
  }
}

// ---------------------------------------------------------------------------
extern "C" void kernel_launch(void* const* d_in, const int* in_sizes, int n_in,
                              void* d_out, int out_size, void* d_ws, size_t ws_size,
                              hipStream_t stream) {
  (void)in_sizes; (void)n_in; (void)out_size; (void)ws_size;
  const float* x    = (const float*)d_in[0];
  const float* Win  = (const float*)d_in[1];
  const float* Wrec = (const float*)d_in[2];
  const float* bias = (const float*)d_in[3];
  float* out = (float*)d_out;

  char* ws = (char*)d_ws;
  bf16_t* Winb   = (bf16_t*)ws;                          // 512 KB
  bf16_t* WrecSw = (bf16_t*)(ws + IN_DIM * UNITS * 2);   // 512 KB

  rnn_init<<<(IN_DIM * UNITS + 255) / 256, 256, 0, stream>>>(Win, Wrec, Winb, WrecSw);
  rnn_gemm_in<<<(ROWS / BM) * (UNITS / BN), 256, 0, stream>>>(x, Winb, bias, out);
  rnn_scan<<<BATCH / 16, 1024, 0, stream>>>(WrecSw, out);
}
